// GlobalDownSample_22436909154641
// MI455X (gfx1250) — matrix operand order repaired and checked by reading, not compiled, not run on hardware
//
#include <hip/hip_runtime.h>

// ---------------------------------------------------------------------------
// GlobalDownSample for MI455X (gfx1250, wave32, WMMA bf16)
//   B=8, C=128, N=4096, M=1024
// v3: software-pipelined k_energy (1 WG/WGP due to 256KB LDS -> in-wave
// latency hiding is mandatory); hardware bf16 converts; load/compute
// reordering in k_qkv. All fragments remain two b128 loads per lane.
// ---------------------------------------------------------------------------

static constexpr int Bc = 8;
static constexpr int Cc = 128;
static constexpr int Nc = 4096;
static constexpr int Mc = 1024;
static constexpr float SCL = 0.08838834764831845f; // 1/sqrt(128)

typedef __attribute__((ext_vector_type(16))) __bf16          v16bf;
typedef __attribute__((ext_vector_type(8)))  float           v8f;
typedef __attribute__((ext_vector_type(8)))  unsigned short  u16x8;

union BF16Frag { v16bf v; u16x8 h[2]; unsigned short u[16]; };

// Native f32->bf16 (RNE): let the backend pick the hardware convert.
__device__ __forceinline__ unsigned short f2bf(float f) {
    __bf16 h = (__bf16)f;
    return __builtin_bit_cast(unsigned short, h);
}

__device__ __forceinline__ v8f wmma_bf16(v16bf a, v16bf b, v8f c) {
    return __builtin_amdgcn_wmma_f32_16x16x32_bf16(
        /*neg_a=*/false, a, /*neg_b=*/false, b,
        /*c_mod=*/(short)0, c, /*reuse_a=*/false, /*reuse_b=*/false);
}

// ---- A fragment (16x32 bf16, §7.12.2). src row-major [m][k], 16B-aligned.
// lane<16: row m, K {0..7}(h0) {16..23}(h1); lane>=16: K {8..15} {24..31}.
__device__ __forceinline__ v16bf load_a_rm(const unsigned short* src, size_t ld) {
    const int lane = threadIdx.x & 31, m = lane & 15, hw = lane >> 4;
    BF16Frag a;
    a.h[0] = *(const u16x8*)(src + (size_t)m * ld + hw * 8);
    a.h[1] = *(const u16x8*)(src + (size_t)m * ld + 16 + hw * 8);
    return a.v;
}

// ---- B fragment (32x16 bf16). Source stored as [n][k] (element (k,n) =
// src[n*ld+k]). lane<16 -> col n, K {0..7} {16..23} ; lane>=16 -> K {8..15} {24..31}.
__device__ __forceinline__ v16bf load_b_nk(const unsigned short* src, size_t ld) {
    const int lane = threadIdx.x & 31, n = lane & 15, hw = lane >> 4;
    BF16Frag b;
    b.h[0] = *(const u16x8*)(src + (size_t)n * ld + hw * 8);
    b.h[1] = *(const u16x8*)(src + (size_t)n * ld + 16 + hw * 8);
    return b.v;
}

// ===========================================================================
// Prep A: weights f32 -> bf16 row-major (96KB total, L2-resident afterwards)
// ===========================================================================
__global__ __launch_bounds__(256)
void k_convw(const float* __restrict__ wq, const float* __restrict__ wk,
             const float* __restrict__ wv, unsigned short* __restrict__ wbf) {
    const int i = blockIdx.x * 256 + threadIdx.x;       // < 3*128*128
    const float* src = (i < Cc * Cc) ? wq : (i < 2 * Cc * Cc) ? wk : wv;
    wbf[i] = f2bf(src[i & (Cc * Cc - 1)]);
}

// ===========================================================================
// Prep B: x f32 [B][C][N] -> bf16 xT [B][N][C] via LDS tile (both global
// sides coalesced). grid = (N/16, B), block = 256.
// ===========================================================================
__global__ __launch_bounds__(256)
void k_xpose(const float* __restrict__ x, unsigned short* __restrict__ xT) {
    __shared__ alignas(16) unsigned short ls[16 * Cc];
    const int n0 = blockIdx.x * 16;
    const int b  = blockIdx.y;
    for (int i = threadIdx.x; i < 16 * Cc; i += 256) {
        const int c = i >> 4, n = i & 15;              // consecutive tid -> consecutive n
        ls[n * Cc + c] = f2bf(x[((size_t)b * Cc + c) * Nc + n0 + n]);
    }
    __syncthreads();
    unsigned short* dst = xT + ((size_t)b * Nc + n0) * Cc;
    for (int i = threadIdx.x; i < 16 * Cc; i += 256)   // consecutive tid -> consecutive c
        dst[i] = ls[i];
}

// ===========================================================================
// Kernel 1: q/k/v projections. One wave owns one 16-point tile and loops all
// 8 output-channel tiles, so the x B-fragments (4 x v16bf) are loaded ONCE.
// q,k stored transposed [B][N][C]; v stored [B][C][N] (combine-friendly).
// grid = (N/16, 3*B), block = 32.
// ===========================================================================
__global__ __launch_bounds__(32)
void k_qkv(const unsigned short* __restrict__ xT,
           const unsigned short* __restrict__ wbf,
           unsigned short* __restrict__ qT, unsigned short* __restrict__ kT,
           unsigned short* __restrict__ vnt) {
    const int n0  = blockIdx.x * 16;
    const int b   = blockIdx.y / 3;
    const int wsd = blockIdx.y % 3;
    const unsigned short* W = wbf + (size_t)wsd * Cc * Cc;
    const size_t bN = (size_t)b * Nc;

    v16bf bx[4];                                      // x tile, reused 8x
#pragma unroll
    for (int kk = 0; kk < 4; ++kk)
        bx[kk] = load_b_nk(xT + (bN + n0) * Cc + kk * 32, Cc);

    const int lane = threadIdx.x & 31, col = lane & 15, hw = lane >> 4;
#pragma unroll
    for (int o0 = 0; o0 < Cc; o0 += 16) {
        // issue all A loads first, then the WMMA chain (scheduler can overlap
        // across the fully-unrolled o-loop)
        v16bf aw[4];
#pragma unroll
        for (int kk = 0; kk < 4; ++kk)
            aw[kk] = load_a_rm(W + (size_t)o0 * Cc + kk * 32, Cc);
        v8f acc = {};
#pragma unroll
        for (int kk = 0; kk < 4; ++kk)
            acc = wmma_bf16(aw[kk], bx[kk], acc);
        // D: row = r + 8*(lane>=16) (= out-ch offset), col = point offset
        if (wsd == 2) {
#pragma unroll
            for (int r = 0; r < 8; ++r)
                vnt[((size_t)b * Cc + o0 + r + 8 * hw) * Nc + n0 + col] = f2bf(acc[r]);
        } else {
            unsigned short* dst = (wsd == 0) ? qT : kT;
#pragma unroll
            for (int r = 0; r < 8; ++r)
                dst[(bN + n0 + col) * Cc + o0 + r + 8 * hw] = f2bf(acc[r]);
        }
    }
}

// ===========================================================================
// Kernel 2: attention statistics. One block = 16 query rows x all 4096 keys.
// Full 16x4096 f32 S row-block lives in LDS (256KB: CDNA5 320KB/WGP), which
// forces 1 WG/WGP -> the key-tile loop is explicitly double-buffered so the
// next tile's b128 loads overlap the current tile's WMMAs.
// grid = (N/16, B), block = 128 (4 waves split the key tiles).
// ===========================================================================
__global__ __launch_bounds__(128, 1)
void k_energy(const unsigned short* __restrict__ qT,
              const unsigned short* __restrict__ kT,
              float* __restrict__ colpart,   // [B][N/16][N]
              float* __restrict__ rowmaxg,   // [B][N]
              float* __restrict__ rowsumg) { // [B][N]
    extern __shared__ float smem[];
    float* S     = smem;               // 16*4096
    float* red   = S + 16 * Nc;        // 128
    float* rmaxl = red + 128;          // 16
    float* rsuml = rmaxl + 16;         // 16

    const int nq0 = blockIdx.x * 16;
    const int b   = blockIdx.y;
    const int tid = threadIdx.x;
    const int wv_ = tid >> 5;
    const size_t bN = (size_t)b * Nc;
    const int lane = tid & 31, col = lane & 15, hw = lane >> 4;

    // A fragments: qT rows nq0..nq0+15, full K=128 (4 frags, reused 64x)
    v16bf aq[4];
#pragma unroll
    for (int kk = 0; kk < 4; ++kk)
        aq[kk] = load_a_rm(qT + (bN + nq0) * Cc + kk * 32, Cc);

    auto ldfrag = [&](v16bf* f, int t) {
        const size_t base = (bN + (size_t)t * 16) * Cc;
        __builtin_prefetch(kT + base + 8 * Cc, 0, 0);   // next tile's lines
#pragma unroll
        for (int kk = 0; kk < 4; ++kk)
            f[kk] = load_b_nk(kT + base + kk * 32, Cc);
    };
    auto compute = [&](const v16bf* f, int t) {
        const int nk0 = t * 16;
        v8f d = {};
#pragma unroll
        for (int kk = 0; kk < 4; ++kk) d = wmma_bf16(aq[kk], f[kk], d);
#pragma unroll
        for (int r = 0; r < 8; ++r)
            S[(r + 8 * hw) * Nc + nk0 + col] = d[r] * SCL;
    };

    // 64 tiles per wave (even) -> clean 2-deep ping-pong pipeline
    v16bf fA[4], fB[4];
    ldfrag(fA, wv_);
    for (int t = wv_; t < Nc / 16; t += 8) {
        ldfrag(fB, t + 4);              // always in range (t+4 <= 255)
        compute(fA, t);
        if (t + 8 < Nc / 16) ldfrag(fA, t + 8);
        compute(fB, t + 4);
    }
    __syncthreads();

    // per-row softmax statistics; S is overwritten with exp(S - rowmax)
    for (int r = 0; r < 16; ++r) {
        float mx = -3.402823466e38f;
        for (int i = tid; i < Nc; i += 128) mx = fmaxf(mx, S[r * Nc + i]);
        red[tid] = mx; __syncthreads();
        for (int s = 64; s > 0; s >>= 1) {
            if (tid < s) red[tid] = fmaxf(red[tid], red[tid + s]);
            __syncthreads();
        }
        const float rm = red[0];
        __syncthreads();
        float sum = 0.f;
        for (int i = tid; i < Nc; i += 128) {
            float e = __expf(S[r * Nc + i] - rm);
            S[r * Nc + i] = e;
            sum += e;
        }
        red[tid] = sum; __syncthreads();
        for (int s = 64; s > 0; s >>= 1) {
            if (tid < s) red[tid] += red[tid + s];
            __syncthreads();
        }
        if (tid == 0) { rmaxl[r] = rm; rsuml[r] = red[0]; }
        __syncthreads();
    }

    if (tid < 16) {
        rowmaxg[bN + nq0 + tid] = rmaxl[tid];
        rowsumg[bN + nq0 + tid] = rsuml[tid];
    }

    // deterministic per-tile partial column sums (no float atomics)
    float inv[16];
#pragma unroll
    for (int r = 0; r < 16; ++r) inv[r] = 1.0f / rsuml[r];
    for (int nk = tid; nk < Nc; nk += 128) {
        float a = 0.f;
#pragma unroll
        for (int r = 0; r < 16; ++r) a += S[r * Nc + nk] * inv[r];
        colpart[((size_t)b * (Nc / 16) + blockIdx.x) * Nc + nk] = a;
    }
}

// Fixed-order reduction of the partial column sums -> selection [B][N]
__global__ __launch_bounds__(256)
void k_colreduce(const float* __restrict__ colpart, float* __restrict__ colsum) {
    const int g = blockIdx.x * blockDim.x + threadIdx.x;  // 0..B*N-1
    const int b = g / Nc, nk = g % Nc;
    float s = 0.f;
    const float* p = colpart + (size_t)b * (Nc / 16) * Nc + nk;
    for (int t = 0; t < Nc / 16; ++t) s += p[(size_t)t * Nc];
    colsum[g] = s;
}

// ===========================================================================
// Kernel 3: exact top-k by rank counting (stable: ties broken by lower index,
// matches jax.lax.top_k). grid = B, block = 256, O(N^2) per batch = trivial.
// ===========================================================================
__global__ __launch_bounds__(256)
void k_topk(const float* __restrict__ colsum, int* __restrict__ idxsel) {
    __shared__ float sel[Nc];
    const int b = blockIdx.x, tid = threadIdx.x;
    for (int i = tid; i < Nc; i += 256) sel[i] = colsum[(size_t)b * Nc + i];
    __syncthreads();
    for (int i = tid; i < Nc; i += 256) {
        const float si = sel[i];
        int rank = 0;
        for (int j = 0; j < Nc; ++j) {
            float sj = sel[j];
            rank += (sj > si) || (sj == si && j < i);
        }
        if (rank < Mc) idxsel[(size_t)b * Mc + rank] = i;
    }
}

// ===========================================================================
// Kernel 4: gather 16 selected rows, RECOMPUTE their attention tiles with
// WMMA (9 GFLOP total -- free vs. re-reading a 537MB matrix), apply stored
// softmax stats, and WMMA against v^T for the [16 x 128] output tile.
// grid = (M/16, B), block = 32 (one wave).
// ===========================================================================
__global__ __launch_bounds__(32)
void k_combine(const unsigned short* __restrict__ qT,
               const unsigned short* __restrict__ kT,
               const unsigned short* __restrict__ vnt,
               const int* __restrict__ idxsel,
               const float* __restrict__ rowmaxg,
               const float* __restrict__ rowsumg,
               float* __restrict__ out) {
    __shared__ int   sidx[16];
    __shared__ float rmaxl[16], rinvl[16];
    __shared__ alignas(16) unsigned short qsel[16 * Cc];  // gathered q rows
    __shared__ alignas(16) unsigned short Pbuf[16 * 32];  // P tile (bf16)

    const int m0 = blockIdx.x * 16;
    const int b  = blockIdx.y;
    const int lane = threadIdx.x & 31, col = lane & 15, hw = lane >> 4;
    const size_t bN = (size_t)b * Nc;

    if (lane < 16) sidx[lane] = idxsel[(size_t)b * Mc + m0 + lane];
    __syncthreads();
    if (lane < 16) {
        rmaxl[lane] = rowmaxg[bN + sidx[lane]];
        rinvl[lane] = 1.0f / rowsumg[bN + sidx[lane]];
    }
    for (int i = lane * 8; i < 16 * Cc; i += 32 * 8) {  // b128 copies
        const int row = i >> 7, c = i & (Cc - 1);
        *(u16x8*)(qsel + i) = *(const u16x8*)(qT + (bN + sidx[row]) * Cc + c);
    }
    __syncthreads();

    v8f acc[8];
    const v8f z = {};
#pragma unroll
    for (int ct = 0; ct < 8; ++ct) acc[ct] = z;

    for (int nk0 = 0; nk0 < Nc; nk0 += 32) {
        if (nk0 + 32 < Nc) {            // stream next iteration's operands
            __builtin_prefetch(kT + (bN + nk0 + 32) * Cc, 0, 0);
            __builtin_prefetch(vnt + ((size_t)b * Cc) * Nc + nk0 + 32, 0, 0);
        }
        // ---- S tile: 16 selected queries x 32 keys, K = 128
        v8f d0 = {}, d1 = {};
#pragma unroll
        for (int kk = 0; kk < Cc; kk += 32) {
            v16bf a = load_a_rm(qsel + kk, Cc);
            d0 = wmma_bf16(a, load_b_nk(kT + (bN + nk0)      * Cc + kk, Cc), d0);
            d1 = wmma_bf16(a, load_b_nk(kT + (bN + nk0 + 16) * Cc + kk, Cc), d1);
        }
        // ---- softmax-ize with stored stats, stage P as bf16 A-tile
#pragma unroll
        for (int r = 0; r < 8; ++r) {
            const int m = r + 8 * hw;
            Pbuf[m * 32 + col]      = f2bf(__expf(d0[r] * SCL - rmaxl[m]) * rinvl[m]);
            Pbuf[m * 32 + 16 + col] = f2bf(__expf(d1[r] * SCL - rmaxl[m]) * rinvl[m]);
        }
        __syncthreads();
        v16bf ap = load_a_rm(Pbuf, 32);
#pragma unroll
        for (int ct = 0; ct < 8; ++ct) {
            // B(k=nk, n=c) = v[b][ct*16+n][nk0+k] : contiguous in k -> b128s
            v16bf bv = load_b_nk(vnt + ((size_t)b * Cc + ct * 16) * Nc + nk0, Nc);
            acc[ct] = wmma_bf16(ap, bv, acc[ct]);
        }
        __syncthreads();
    }

    // out[b][c][m]: D row = selected-row offset, col = channel offset
#pragma unroll
    for (int ct = 0; ct < 8; ++ct)
#pragma unroll
        for (int r = 0; r < 8; ++r)
            out[((size_t)b * Cc + ct * 16 + col) * Mc + m0 + r + 8 * hw] = acc[ct][r];
}

// ===========================================================================
extern "C" void kernel_launch(void* const* d_in, const int* in_sizes, int n_in,
                              void* d_out, int out_size, void* d_ws, size_t ws_size,
                              hipStream_t stream) {
    const float* x  = (const float*)d_in[0];
    const float* wq = (const float*)d_in[1];
    const float* wk = (const float*)d_in[2];
    const float* wv = (const float*)d_in[3];
    float* out = (float*)d_out;

    char* ws = (char*)d_ws;
    size_t off = 0;
    unsigned short* wbf = (unsigned short*)(ws + off); off += (size_t)3 * Cc * Cc * 2;
    unsigned short* xT  = (unsigned short*)(ws + off); off += (size_t)Bc * Nc * Cc * 2;
    unsigned short* qT  = (unsigned short*)(ws + off); off += (size_t)Bc * Nc * Cc * 2;
    unsigned short* kT  = (unsigned short*)(ws + off); off += (size_t)Bc * Nc * Cc * 2;
    unsigned short* vnt = (unsigned short*)(ws + off); off += (size_t)Bc * Nc * Cc * 2;
    float* colpart = (float*)(ws + off); off += (size_t)Bc * (Nc / 16) * Nc * 4;
    float* colsum  = (float*)(ws + off); off += (size_t)Bc * Nc * 4;
    float* rowmaxg = (float*)(ws + off); off += (size_t)Bc * Nc * 4;
    float* rowsumg = (float*)(ws + off); off += (size_t)Bc * Nc * 4;
    int*   idxsel  = (int*)  (ws + off); off += (size_t)Bc * Mc * 4;

    // 0) one-time bf16 conversion: weights (row-major) + x (transposed)
    k_convw<<<(3 * Cc * Cc) / 256, 256, 0, stream>>>(wq, wk, wv, wbf);
    k_xpose<<<dim3(Nc / 16, Bc), 256, 0, stream>>>(x, xT);

    // 1) q/k/v projection (WMMA bf16); q,k -> [B][N][C], v -> [B][C][N]
    k_qkv<<<dim3(Nc / 16, 3 * Bc), 32, 0, stream>>>(xT, wbf, qT, kT, vnt);

    // 2) softmax stats + partial column sums; 16x4096 f32 S-block in LDS
    const size_t smem2 = (size_t)(16 * Nc + 128 + 32) * sizeof(float); // ~263KB
    k_energy<<<dim3(Nc / 16, Bc), 128, smem2, stream>>>(qT, kT, colpart, rowmaxg, rowsumg);

    // 3) deterministic column-sum reduction + exact stable top-k
    k_colreduce<<<(Bc * Nc) / 256, 256, 0, stream>>>(colpart, colsum);
    k_topk<<<Bc, 256, 0, stream>>>(colsum, idxsel);

    // 4) recompute selected rows + weighted-V combine (WMMA bf16)
    k_combine<<<dim3(Mc / 16, Bc), 32, 0, stream>>>(qT, kT, vnt, idxsel, rowmaxg, rowsumg, out);
}
